// LorentzAttention_10969346474198
// MI455X (gfx1250) — matrix operand order repaired and checked by reading, not compiled, not run on hardware
//
#include <hip/hip_runtime.h>
#include <hip/hip_bf16.h>
#include <math.h>

// ---------------------------------------------------------------------------
// Lorentz attention, fully WMMA-based for gfx1250 (CDNA5, wave32),
// with TDM (tensor_load_to_lds) staging of K/V tiles in the attention kernel.
// B=2, T=2048, D=1024, H=16, HD=64.
// ---------------------------------------------------------------------------

typedef __attribute__((ext_vector_type(16))) _Float16     v16h;
typedef __attribute__((ext_vector_type(8)))  _Float16     v8h;
typedef __attribute__((ext_vector_type(8)))  float        v8f;
typedef __attribute__((ext_vector_type(4)))  unsigned int u32x4;
typedef __attribute__((ext_vector_type(8)))  int          i32x8;
typedef __attribute__((ext_vector_type(4)))  int          i32x4;

static constexpr int  Bc  = 2;
static constexpr int  Tc  = 2048;
static constexpr int  Dc  = 1024;
static constexpr int  Hc  = 16;
static constexpr int  HDc = 64;
static constexpr int  TOK = Bc * Tc;          // 4096 tokens

#if __has_builtin(__builtin_amdgcn_tensor_load_to_lds) && \
    __has_builtin(__builtin_amdgcn_s_wait_tensorcnt)
#define USE_TDM 1
#else
#define USE_TDM 0
#endif

__device__ inline v8f wmma_f16(v16h a, v16h b, v8f c) {
  // D = A(16x32 f16) * B(32x16 f16) + C(16x16 f32)
  return __builtin_amdgcn_wmma_f32_16x16x32_f16(
      false, a, false, b, (short)0, c, false, false);
}

// A-fragment (16x32 f16): lane r=lane&15 holds row M=r; halves 0..15 = K 2s..2s+15,
// s = 8*(lane>=16).
__device__ inline v16h ldA(const _Float16* rowk, int s) {
  v8h lo = *(const v8h*)(rowk + 2 * s);
  v8h hv = *(const v8h*)(rowk + 2 * s + 8);
  v16h a;
#pragma unroll
  for (int i = 0; i < 8; ++i) { a[i] = lo[i]; a[8 + i] = hv[i]; }
  return a;
}

__device__ inline float redmax16(float v) {
#pragma unroll
  for (int m = 1; m < 16; m <<= 1) v = fmaxf(v, __shfl_xor(v, m, 32));
  return v;
}
__device__ inline float redsum16(float v) {
#pragma unroll
  for (int m = 1; m < 16; m <<= 1) v += __shfl_xor(v, m, 32);
  return v;
}

#if USE_TDM
// Issue a 2D TDM tile load: tile_d1 rows of tile_d0 elements (f16), row
// stride stride0 (elements), packed contiguously into LDS at lds_off.
// D# bitfields per CDNA5 ISA ch. 7/8 (group0: count/lds/global/type,
// group1: data_size=2B, dims, strides).  tensor dims == tile dims so no
// OOB clamping applies.
__device__ inline void tdm_load_2d(unsigned lds_off, unsigned long long gaddr,
                                   unsigned tile_d0, unsigned tile_d1,
                                   unsigned long long stride0) {
  u32x4 g0;
  g0[0] = 1u;                                        // count=1, no gather
  g0[1] = lds_off;                                   // LDS byte address
  g0[2] = (unsigned)(gaddr & 0xffffffffull);         // global_addr[31:0]
  g0[3] = (unsigned)((gaddr >> 32) & 0x01ffffffull)  // global_addr[56:32]
        | (2u << 30);                                // type = 2 ("image")
  i32x8 g1;
  g1[0] = (int)(1u << 16);                           // data_size = 1 (2 bytes)
  g1[1] = (int)((tile_d0 & 0xffffu) << 16);          // tensor_dim0[15:0]
  g1[2] = (int)(((tile_d0 >> 16) & 0xffffu) |        // tensor_dim0[31:16]
                ((tile_d1 & 0xffffu) << 16));        // tensor_dim1[15:0]
  g1[3] = (int)(((tile_d1 >> 16) & 0xffffu) |        // tensor_dim1[31:16]
                ((tile_d0 & 0xffffu) << 16));        // tile_dim0
  g1[4] = (int)(tile_d1 & 0xffffu);                  // tile_dim1 (tile_dim2=0)
  g1[5] = (int)(unsigned)(stride0 & 0xffffffffull);  // dim0_stride[31:0]
  g1[6] = (int)(unsigned)((stride0 >> 32) & 0xffffull); // dim0_stride[47:32]
  g1[7] = 0;
  i32x4 g2 = {0, 0, 0, 0};
  i32x4 g3 = {0, 0, 0, 0};
#if defined(__clang_major__) && (__clang_major__ >= 23)
  i32x8 gx = {0, 0, 0, 0, 0, 0, 0, 0};
  __builtin_amdgcn_tensor_load_to_lds(g0, g1, g2, g3, gx, 0);
#else
  __builtin_amdgcn_tensor_load_to_lds(g0, g1, g2, g3, 0);
#endif
}
#endif  // USE_TDM

// ---------------------------------------------------------------------------
// f32 -> f16 conversion (weights)
// ---------------------------------------------------------------------------
__global__ void cvt16_kernel(const float* __restrict__ s, _Float16* __restrict__ d, int n) {
  int i = blockIdx.x * 256 + threadIdx.x;
  if (i < n) d[i] = (_Float16)s[i];
}

// ---------------------------------------------------------------------------
// log_map0: h_tan = asinh(sqrt(c)*||h||)/(sqrt(c)*||h||) * h   (per token row)
// ---------------------------------------------------------------------------
__global__ void logmap_kernel(const float* __restrict__ h, const float* __restrict__ curv,
                              _Float16* __restrict__ htan) {
  const int row = blockIdx.x;
  const float* hr = h + (size_t)row * Dc;
  float ss = 0.f;
  for (int i = threadIdx.x; i < Dc; i += 256) { float x = hr[i]; ss += x * x; }
#pragma unroll
  for (int m = 1; m < 32; m <<= 1) ss += __shfl_xor(ss, m, 32);
  __shared__ float red[8];
  if ((threadIdx.x & 31) == 0) red[threadIdx.x >> 5] = ss;
  __syncthreads();
  float tot = 0.f;
#pragma unroll
  for (int w = 0; w < 8; ++w) tot += red[w];
  const float c  = *curv;
  const float nc = sqrtf(c * tot);                 // sqrt(c)*||h||
  const float coef = (nc > 1e-7f) ? (asinhf(nc) / nc) : 1.0f;
  _Float16* orow = htan + (size_t)row * Dc;
  for (int i = threadIdx.x; i < Dc; i += 256) orow[i] = (_Float16)(hr[i] * coef);
}

// ---------------------------------------------------------------------------
// GEMM: C[m,n] = sum_k A[m,k]*W[n,k] + bias[n]  (y = x @ W^T + b)
// One wave computes a 64x64 tile (4x4 WMMA sub-tiles): 16 WMMAs per K-step,
// 4x fragment reuse in each direction -> ~16 B/lane/WMMA of load traffic.
// ---------------------------------------------------------------------------
__global__ __launch_bounds__(32) void gemm64_kernel(
    const _Float16* __restrict__ A, const _Float16* __restrict__ W,
    const float* __restrict__ bias, float* __restrict__ C, int M, int N, int K) {
  const int lane = threadIdx.x;
  const int r = lane & 15, hi = lane >> 4, s = hi * 8;
  const int m0 = blockIdx.y * 64, n0 = blockIdx.x * 64;
  const _Float16* ar[4];
  const _Float16* br[4];
#pragma unroll
  for (int i = 0; i < 4; ++i) ar[i] = A + (size_t)(m0 + i * 16 + r) * K;
#pragma unroll
  for (int j = 0; j < 4; ++j) br[j] = W + (size_t)(n0 + j * 16 + r) * K + hi * 16;

  v8f acc[4][4] = {};
  for (int k = 0; k < K; k += 32) {
    v16h a[4], b[4];
#pragma unroll
    for (int i = 0; i < 4; ++i) a[i] = ldA(ar[i] + k, s);
#pragma unroll
    for (int j = 0; j < 4; ++j) b[j] = *(const v16h*)(br[j] + k);
#pragma unroll
    for (int i = 0; i < 4; ++i)
#pragma unroll
      for (int j = 0; j < 4; ++j) acc[i][j] = wmma_f16(a[i], b[j], acc[i][j]);
  }
#pragma unroll
  for (int j = 0; j < 4; ++j) {
    const float bn = bias[n0 + j * 16 + r];
#pragma unroll
    for (int i = 0; i < 4; ++i)
#pragma unroll
      for (int t = 0; t < 8; ++t)   // C row = t + 8*hi, col = r
        C[(size_t)(m0 + i * 16 + t + hi * 8) * N + n0 + j * 16 + r] =
            acc[i][j][t] + bn;
  }
}

// ---------------------------------------------------------------------------
// exp_map0 per head-row of 64 + time component.
// grid = B*H*T blocks of 32; block idx = (b*H + h)*T + t.
// ---------------------------------------------------------------------------
__global__ __launch_bounds__(32) void expmap_kernel(
    const float* __restrict__ tanb, const float* __restrict__ curv,
    _Float16* __restrict__ hyp, float* __restrict__ timeb) {
  const int idx  = blockIdx.x;
  const int t    = idx & (Tc - 1);
  const int bh   = idx >> 11;                    // / Tc
  const int hh   = bh & (Hc - 1);
  const int bb   = bh >> 4;
  const int lane = threadIdx.x;
  const float* src = tanb + (size_t)(bb * Tc + t) * Dc + hh * HDc;
  const float x0 = src[lane], x1 = src[lane + 32];
  float ss = x0 * x0 + x1 * x1;
#pragma unroll
  for (int m = 1; m < 32; m <<= 1) ss += __shfl_xor(ss, m, 32);
  const float c  = *curv;
  const float nc = sqrtf(c * ss);
  const float coef = (nc > 1e-7f) ? (sinhf(nc) / nc) : 1.0f;
  _Float16* dst = hyp + ((size_t)bh * Tc + t) * HDc;
  dst[lane]      = (_Float16)(x0 * coef);
  dst[lane + 32] = (_Float16)(x1 * coef);
  if (lane == 0)
    timeb[(size_t)bh * Tc + t] = sqrtf(1.0f / c + coef * coef * ss);
}

// ---------------------------------------------------------------------------
// V transpose: (B,T,D) f32 -> (B,H,HD,T) f16 so PV B-fragments are contiguous.
// ---------------------------------------------------------------------------
__global__ void vtrans_kernel(const float* __restrict__ vtan, _Float16* __restrict__ vT) {
  const size_t idx = (size_t)blockIdx.x * 256 + threadIdx.x;   // over B*T*D
  const int col = (int)(idx & (Dc - 1));
  const size_t tok = idx >> 10;
  const int t = (int)(tok & (Tc - 1));
  const int b = (int)(tok >> 11);
  const int hh = col >> 6, hd = col & (HDc - 1);
  vT[((size_t)(b * Hc + hh) * HDc + hd) * Tc + t] = (_Float16)vtan[idx];
}

// ---------------------------------------------------------------------------
// Flash attention over full T=2048 with Lorentz logits:
//   logits = (q_time*k_time - q_hyp.k_hyp) / sqrt(HD)
// One wave per 16-query tile per (b,h).  Keys consumed 32 at a time.
// K/V tiles staged into LDS by the Tensor Data Mover, double buffered.
// ---------------------------------------------------------------------------
__global__ __launch_bounds__(32) void attn_kernel(
    const _Float16* __restrict__ Qh, const _Float16* __restrict__ Kh,
    const _Float16* __restrict__ Vt, const float* __restrict__ qtime,
    const float* __restrict__ ktime, _Float16* __restrict__ Oc) {
  const int lane = threadIdx.x;
  const int r = lane & 15, hi = lane >> 4, s = hi * 8;
  const int bh = blockIdx.y;                 // 0..B*H-1
  const int q0 = blockIdx.x * 16;

  // Q fragments held in registers for the whole key loop.
  const _Float16* qrow = Qh + ((size_t)bh * Tc + q0 + r) * HDc;
  const v16h aq0 = ldA(qrow, s);             // hd 0..31
  const v16h aq1 = ldA(qrow + 32, s);        // hd 32..63

  float qtv[8];
#pragma unroll
  for (int j = 0; j < 8; ++j)
    qtv[j] = qtime[(size_t)bh * Tc + q0 + j + hi * 8];

  float mrow[8], lrow[8];
  v8f Oacc[4] = {};
#pragma unroll
  for (int j = 0; j < 8; ++j) { mrow[j] = -3.0e38f; lrow[j] = 0.f; }

  __shared__ _Float16 P[16 * 32];            // P tile: C-layout -> A-layout
  const _Float16* Kb  = Kh + (size_t)bh * Tc * HDc;   // (T,HD)
  const _Float16* Vb  = Vt + (size_t)bh * HDc * Tc;   // (HD,T)
  const float*    ktb = ktime + (size_t)bh * Tc;

#if USE_TDM
  __shared__ _Float16 Kt[2][32 * 64];        // key-major (key, hd)
  __shared__ _Float16 Vl[2][64 * 32];        // hd-major  (hd, key)
  // Prologue: stage chunk 0.
  tdm_load_2d((unsigned)(size_t)&Kt[0][0],
              (unsigned long long)(size_t)Kb, 64u, 32u, 64ull);
  tdm_load_2d((unsigned)(size_t)&Vl[0][0],
              (unsigned long long)(size_t)Vb, 32u, 64u, (unsigned long long)Tc);
#endif

  for (int kb = 0; kb < Tc; kb += 32) {
#if USE_TDM
    const int cur = (kb >> 5) & 1;
    if (kb + 32 < Tc) {
      // Prefetch next chunk into the other buffer, then wait for current.
      tdm_load_2d((unsigned)(size_t)&Kt[cur ^ 1][0],
                  (unsigned long long)(size_t)(Kb + (size_t)(kb + 32) * HDc),
                  64u, 32u, 64ull);
      tdm_load_2d((unsigned)(size_t)&Vl[cur ^ 1][0],
                  (unsigned long long)(size_t)(Vb + kb + 32),
                  32u, 64u, (unsigned long long)Tc);
      __builtin_amdgcn_s_wait_tensorcnt((short)2);
    } else {
      __builtin_amdgcn_s_wait_tensorcnt((short)0);
    }
    const _Float16* kt = &Kt[cur][0];
    const _Float16* vt = &Vl[cur][0];
    const _Float16* k0p = kt + (size_t)r * HDc + hi * 16;
    const _Float16* k1p = kt + (size_t)(16 + r) * HDc + hi * 16;
#else
    const _Float16* k0p = Kb + (size_t)(kb + r) * HDc + hi * 16;
    const _Float16* k1p = Kb + (size_t)(kb + 16 + r) * HDc + hi * 16;
#endif
    // --- S = Q @ K^T for 32 keys (two 16x16 tiles) ---
    v8f s0 = {}, s1 = {};
    s0 = wmma_f16(aq0, *(const v16h*)(k0p), s0);
    s0 = wmma_f16(aq1, *(const v16h*)(k0p + 32), s0);
    s1 = wmma_f16(aq0, *(const v16h*)(k1p), s1);
    s1 = wmma_f16(aq1, *(const v16h*)(k1p + 32), s1);

    const float kt0 = ktb[kb + r];
    const float kt1 = ktb[kb + 16 + r];

    float p0[8], p1[8], scl[8];
#pragma unroll
    for (int j = 0; j < 8; ++j) {
      const float l0 = (qtv[j] * kt0 - s0[j]) * 0.125f;   // 1/sqrt(64)
      const float l1 = (qtv[j] * kt1 - s1[j]) * 0.125f;
      const float mx = redmax16(fmaxf(l0, l1));
      const float mn = fmaxf(mrow[j], mx);
      scl[j]  = __expf(mrow[j] - mn);
      mrow[j] = mn;
      p0[j] = __expf(l0 - mn);
      p1[j] = __expf(l1 - mn);
      lrow[j] = lrow[j] * scl[j] + redsum16(p0[j] + p1[j]);
    }
#pragma unroll
    for (int j = 0; j < 8; ++j) {
      Oacc[0][j] *= scl[j]; Oacc[1][j] *= scl[j];
      Oacc[2][j] *= scl[j]; Oacc[3][j] *= scl[j];
    }

    // --- C-layout P -> LDS -> A-layout fragment ---
    __syncthreads();   // protect prior iteration's LDS reads
#pragma unroll
    for (int j = 0; j < 8; ++j) {
      P[(j + hi * 8) * 32 + r]      = (_Float16)p0[j];
      P[(j + hi * 8) * 32 + 16 + r] = (_Float16)p1[j];
    }
    __syncthreads();
    const v16h pa = ldA(&P[r * 32], s);      // 16x32 A-frag of probabilities

    // --- O += P @ V ---
#pragma unroll
    for (int c = 0; c < 4; ++c) {
#if USE_TDM
      const _Float16* vp = vt + (size_t)(c * 16 + r) * 32 + hi * 16;
#else
      const _Float16* vp = Vb + (size_t)(c * 16 + r) * Tc + kb + hi * 16;
#endif
      Oacc[c] = wmma_f16(pa, *(const v16h*)vp, Oacc[c]);
    }
  }

  // normalize and write to (B,T,D) f16 for the output projection
  const int b = bh >> 4, hh = bh & (Hc - 1);
#pragma unroll
  for (int j = 0; j < 8; ++j) {
    const float inv = 1.0f / lrow[j];
    const size_t base =
        ((size_t)(b * Tc + q0 + j + hi * 8)) * Dc + hh * HDc + r;
    Oc[base]      = (_Float16)(Oacc[0][j] * inv);
    Oc[base + 16] = (_Float16)(Oacc[1][j] * inv);
    Oc[base + 32] = (_Float16)(Oacc[2][j] * inv);
    Oc[base + 48] = (_Float16)(Oacc[3][j] * inv);
  }
}

// ---------------------------------------------------------------------------
// Launch
// ---------------------------------------------------------------------------
extern "C" void kernel_launch(void* const* d_in, const int* in_sizes, int n_in,
                              void* d_out, int out_size, void* d_ws, size_t ws_size,
                              hipStream_t stream) {
  (void)in_sizes; (void)n_in; (void)out_size; (void)ws_size;

  const float* h    = (const float*)d_in[0];
  const float* curv = (const float*)d_in[1];
  const float* wq_w = (const float*)d_in[2];
  const float* wq_b = (const float*)d_in[3];
  const float* wk_w = (const float*)d_in[4];
  const float* wk_b = (const float*)d_in[5];
  const float* wv_w = (const float*)d_in[6];
  const float* wv_b = (const float*)d_in[7];
  const float* wo_w = (const float*)d_in[8];
  const float* wo_b = (const float*)d_in[9];
  float* out = (float*)d_out;

  char* ws = (char*)d_ws;
  // Workspace map (~88.5 MB):
  _Float16* htan = (_Float16*)(ws);                       //  8 MB  (4096x1024 f16)
  _Float16* wq16 = (_Float16*)(ws + (8ull  << 20));       //  2 MB each
  _Float16* wk16 = wq16 + (size_t)Dc * Dc;
  _Float16* wv16 = wk16 + (size_t)Dc * Dc;
  _Float16* wo16 = wv16 + (size_t)Dc * Dc;                // ends at 16 MB
  float* qtan = (float*)(ws + (16ull << 20));             // 16 MB
  float* ktan = (float*)(ws + (32ull << 20));             // 16 MB
  float* vtan = (float*)(ws + (48ull << 20));             // 16 MB -> ends 64 MB
  _Float16* qhyp = (_Float16*)(ws + (64ull << 20));       //  8 MB (B,H,T,HD)
  _Float16* khyp = (_Float16*)(ws + (72ull << 20));       //  8 MB
  _Float16* vT   = (_Float16*)(ws + (80ull << 20));       //  8 MB (B,H,HD,T)
  float* qtime = (float*)(ws + (88ull << 20));            // 256 KB (B,H,T)
  float* ktime = qtime + (size_t)Bc * Hc * Tc;
  _Float16* attn16 = (_Float16*)qtan;                     // reuse qtan region

  const int NW = Dc * Dc;                                 // 1M weights each
  cvt16_kernel<<<NW / 256, 256, 0, stream>>>(wq_w, wq16, NW);
  cvt16_kernel<<<NW / 256, 256, 0, stream>>>(wk_w, wk16, NW);
  cvt16_kernel<<<NW / 256, 256, 0, stream>>>(wv_w, wv16, NW);
  cvt16_kernel<<<NW / 256, 256, 0, stream>>>(wo_w, wo16, NW);

  logmap_kernel<<<TOK, 256, 0, stream>>>(h, curv, htan);

  dim3 gg(Dc / 64, TOK / 64);                             // (16, 64) 64x64 tiles
  gemm64_kernel<<<gg, 32, 0, stream>>>(htan, wq16, wq_b, qtan, TOK, Dc, Dc);
  gemm64_kernel<<<gg, 32, 0, stream>>>(htan, wk16, wk_b, ktan, TOK, Dc, Dc);
  gemm64_kernel<<<gg, 32, 0, stream>>>(htan, wv16, wv_b, vtan, TOK, Dc, Dc);

  const int ROWS = Bc * Hc * Tc;                          // 65536
  expmap_kernel<<<ROWS, 32, 0, stream>>>(qtan, curv, qhyp, qtime);
  expmap_kernel<<<ROWS, 32, 0, stream>>>(ktan, curv, khyp, ktime);
  vtrans_kernel<<<(TOK * Dc) / 256, 256, 0, stream>>>(vtan, vT);

  attn_kernel<<<dim3(Tc / 16, Bc * Hc), 32, 0, stream>>>(
      qhyp, khyp, vT, qtime, ktime, attn16);

  gemm64_kernel<<<gg, 32, 0, stream>>>(attn16, wo16, wo_b, out, TOK, Dc, Dc);
}
